// KerasSelfAttention_69767448756609
// MI455X (gfx1250) — matrix operand order repaired and checked by reading, not compiled, not run on hardware
//
#include <hip/hip_runtime.h>

// ---------------------------------------------------------------------------
// KerasSelfAttention (B=4, T=4096, C=64), fp32 in/out.
//   q = xWq+bq ; k = xWk+bk ; v = xWv+bv
//   attn = softmax(k q^T) v ;  out = gamma*attn + x
// Compute-bound (17.2 GFLOP vs ~8MB HBM): f16 WMMA flash attention with
// f32 accumulation; q/k/vT working set (6MB f16) is L2-resident.
// Row-sums of P are computed on the matrix pipe (P @ ones) instead of
// lane shuffles, keeping VALU for exp/max only.
// ---------------------------------------------------------------------------

typedef __attribute__((ext_vector_type(16))) _Float16 v16h;
typedef __attribute__((ext_vector_type(8)))  _Float16 v8h;
typedef __attribute__((ext_vector_type(4)))  _Float16 v4h;
typedef __attribute__((ext_vector_type(8)))  float    v8f;

#define WMMA_F16(A, B, C) \
  __builtin_amdgcn_wmma_f32_16x16x32_f16(false, (A), false, (B), (short)0, (C), false, false)

constexpr int Tn = 4096;
constexpr int Cn = 64;
constexpr int Bn = 4;

// Build a 16-half operand register from two 16-byte chunks (used here for A and B alike: per-lane
// 16 contiguous K halfs [16h,16h+16) via ld16c).
static __device__ __forceinline__ v16h ld16(const _Float16* p0, const _Float16* p1) {
  union { v16h v; v8h h[2]; } u;
  u.h[0] = *(const v8h*)p0;
  u.h[1] = *(const v8h*)p1;
  return u.v;
}
static __device__ __forceinline__ v16h ld16c(const _Float16* p) { return ld16(p, p + 8); }

// Max-reduction across each 16-lane half of the wave32 (C-layout: a matrix
// row's 16 N-elements live in lanes 0-15 or 16-31 of one VGPR).
static __device__ __forceinline__ float half_max(float v) {
  v = fmaxf(v, __shfl_xor(v, 1, 32));
  v = fmaxf(v, __shfl_xor(v, 2, 32));
  v = fmaxf(v, __shfl_xor(v, 4, 32));
  v = fmaxf(v, __shfl_xor(v, 8, 32));
  return v;
}

// ------------------------------ prep kernels -------------------------------

// x (f32) -> xh (f16), 4 elements/thread: b128 load, b64 store.
__global__ void prep_convert_x(const float* __restrict__ x, _Float16* __restrict__ xh, int n4) {
  int i = blockIdx.x * blockDim.x + threadIdx.x;
  if (i < n4) {
    const float4 f = ((const float4*)x)[i];
    union { v4h v; _Float16 e[4]; } u;
    u.e[0] = (_Float16)f.x; u.e[1] = (_Float16)f.y;
    u.e[2] = (_Float16)f.z; u.e[3] = (_Float16)f.w;
    ((v4h*)xh)[i] = u.v;
  }
}

// WtAll[mat][d][c] = W[mat][c][d]  (f16) so B-operand lanes read contiguous.
__global__ void prep_transpose_w(const float* __restrict__ Wq, const float* __restrict__ Wk,
                                 const float* __restrict__ Wv, _Float16* __restrict__ WtAll) {
  int i = blockIdx.x * blockDim.x + threadIdx.x;
  if (i >= 3 * Cn * Cn) return;
  int mat = i / (Cn * Cn), rem = i % (Cn * Cn);
  int d = rem / Cn, c = rem % Cn;
  const float* W = (mat == 0) ? Wq : (mat == 1) ? Wk : Wv;
  WtAll[i] = (_Float16)W[c * Cn + d];
}

// --------------------------- projection (WMMA) -----------------------------
// grid (B*T/16, 3), block 32. One wave: 16 rows x 64 cols of one projection.
// q,k stored row-major f16; v stored transposed [B][64][T] f16 for the P@V gemm.

__global__ void __launch_bounds__(32)
proj_wmma(const _Float16* __restrict__ xh, const _Float16* __restrict__ WtAll,
          const float* __restrict__ bq, const float* __restrict__ bk,
          const float* __restrict__ bv,
          _Float16* __restrict__ qh, _Float16* __restrict__ kh, _Float16* __restrict__ vT) {
  const int lane = threadIdx.x;
  const int lm  = lane & 15;
  const int hw  = lane >> 4;
  const int lk8 = hw * 8;    // unused (A and B both use klo)
  const int klo = hw * 16;   // B-layout per-lane K base
  const int row0 = blockIdx.x * 16;
  const int mat  = blockIdx.y;

  const _Float16* Wt = WtAll + (size_t)mat * Cn * Cn;
  const float* bias = (mat == 0) ? bq : (mat == 1) ? bk : bv;

  const _Float16* xrow = xh + (size_t)(row0 + lm) * Cn;
  v16h A0 = ld16c(xrow + klo);      // K = c 0..31
  v16h A1 = ld16c(xrow + 32 + klo); // K = c 32..63

#pragma unroll
  for (int n0 = 0; n0 < 4; ++n0) {
    const _Float16* wrow = Wt + (size_t)(16 * n0 + lm) * Cn; // row d of W^T
    v16h B0 = ld16c(wrow + klo);
    v16h B1 = ld16c(wrow + 32 + klo);
    float bval = bias[16 * n0 + lm];
    v8f acc;
#pragma unroll
    for (int r = 0; r < 8; ++r) acc[r] = bval;     // bias pre-loaded into C
    acc = WMMA_F16(A1, B1, WMMA_F16(A0, B0, acc));
    if (mat < 2) {
      _Float16* dst = (mat == 0) ? qh : kh;
#pragma unroll
      for (int r = 0; r < 8; ++r)
        dst[(size_t)(row0 + r + 8 * hw) * Cn + 16 * n0 + lm] = (_Float16)acc[r];
    } else {
      // vT[b][d][t]: batch is block-uniform, 8 per-lane elements contiguous
      // in t -> single b128 store per n0 tile.
      union { v8h v; _Float16 e[8]; } pk;
#pragma unroll
      for (int r = 0; r < 8; ++r) pk.e[r] = (_Float16)acc[r];
      const int bb = row0 >> 12;
      const int tt = (row0 & (Tn - 1)) + 8 * hw;
      *(v8h*)(vT + ((size_t)(bb * Cn) + 16 * n0 + lm) * Tn + tt) = pk.v;
    }
  }
}

// --------------------------- flash attention -------------------------------
// grid B*T/16, block 32. One wave owns 16 "t" rows (A-operand = k rows),
// streams s in steps of 32 (two 16x16 score tiles), online softmax,
// P routed through LDS (C-layout -> A-layout), P@V via vT rows.
// 9 WMMA / iteration: 4 scores + 4 P@V + 1 row-sum (P @ ones).

__global__ void __launch_bounds__(32)
flash_attn(const _Float16* __restrict__ kh, const _Float16* __restrict__ qh,
           const _Float16* __restrict__ vT, const float* __restrict__ xin,
           const float* __restrict__ gamma, float* __restrict__ out) {
  __shared__ __align__(16) _Float16 Pbuf[16 * 32];

  const int lane = threadIdx.x;
  const int lm  = lane & 15;
  const int hw  = lane >> 4;
  const int lk8 = hw * 8;
  const int klo = hw * 16;
  const int tile = blockIdx.x;
  const int b  = tile >> 8;                 // 256 tiles per batch
  const int t0 = (tile & 255) << 4;

  const _Float16* kb = kh + (size_t)(b * Tn + t0) * Cn;
  const _Float16* qb = qh + (size_t)b * Tn * Cn;
  const _Float16* vb = vT + (size_t)b * Cn * Tn;

  // A operands: k rows (resident for whole s-loop)
  const _Float16* krow = kb + (size_t)lm * Cn;
  v16h Ak0 = ld16c(krow + klo);
  v16h Ak1 = ld16c(krow + 32 + klo);

  // All-ones B operand: rowsum(P) = P @ ones on the matrix pipe.
  v16h ones;
#pragma unroll
  for (int i = 0; i < 16; ++i) ones[i] = (_Float16)1.0f;

  v8f acc[4];
  float mst[8], lst[8], scl[8];
  v8f zero = {0.f, 0.f, 0.f, 0.f, 0.f, 0.f, 0.f, 0.f};
#pragma unroll
  for (int i = 0; i < 4; ++i) acc[i] = zero;
#pragma unroll
  for (int r = 0; r < 8; ++r) { mst[r] = -3.0e38f; lst[r] = 0.f; }

  for (int s0 = 0; s0 < Tn; s0 += 32) {
    // B operands from q rows (column n of q^T == row n of q: contiguous)
    const _Float16* qa = qb + (size_t)(s0 + lm) * Cn;
    const _Float16* qc = qb + (size_t)(s0 + 16 + lm) * Cn;
    v16h Bqa0 = ld16c(qa + klo);
    v16h Bqa1 = ld16c(qa + 32 + klo);
    v16h Bqb0 = ld16c(qc + klo);
    v16h Bqb1 = ld16c(qc + 32 + klo);
    if (s0 + 32 < Tn) {  // warm caches for next step (global_prefetch_b8)
      __builtin_prefetch(qb + (size_t)(s0 + 32 + lm) * Cn, 0, 1);
      __builtin_prefetch(vb + (size_t)lm * Tn + s0 + 32, 0, 1);
    }

    v8f Sa = WMMA_F16(Ak1, Bqa1, WMMA_F16(Ak0, Bqa0, zero)); // s0..s0+15
    v8f Sb = WMMA_F16(Ak1, Bqb1, WMMA_F16(Ak0, Bqb0, zero)); // s0+16..s0+31

    // Online softmax per row (rows striped: vgpr r, half hw -> row r+8*hw)
#pragma unroll
    for (int r = 0; r < 8; ++r) {
      float mx   = half_max(fmaxf(Sa[r], Sb[r]));
      float mnew = fmaxf(mst[r], mx);
      scl[r]     = __expf(mst[r] - mnew);
      mst[r]     = mnew;
      float pa   = __expf(Sa[r] - mnew);
      float pb   = __expf(Sb[r] - mnew);
#pragma unroll
      for (int ct = 0; ct < 4; ++ct) acc[ct][r] *= scl[r];
      Pbuf[(r + 8 * hw) * 32 + lm]      = (_Float16)pa;
      Pbuf[(r + 8 * hw) * 32 + 16 + lm] = (_Float16)pb;
    }
    __syncthreads();

    // Reload P in A-layout; row-sum via matrix pipe; accumulate P@V.
    const _Float16* prow = Pbuf + lm * 32;
    v16h Ap = ld16c(prow + klo);
    v8f rs = WMMA_F16(Ap, ones, zero);   // rs[r] = rowsum(P) broadcast over N
#pragma unroll
    for (int r = 0; r < 8; ++r) lst[r] = lst[r] * scl[r] + rs[r];
#pragma unroll
    for (int ct = 0; ct < 4; ++ct) {
      v16h Bv = ld16c(vb + (size_t)(16 * ct + lm) * Tn + s0 + klo);
      acc[ct] = WMMA_F16(Ap, Bv, acc[ct]);
    }
    __syncthreads();
  }

  // Epilogue: out = gamma * attn / l + x
  const float g = gamma[0];
  const float* xrow = xin + (size_t)(b * Tn + t0) * Cn;
  float* orow = out + (size_t)(b * Tn + t0) * Cn;
#pragma unroll
  for (int r = 0; r < 8; ++r) {
    const int m = r + 8 * hw;
    const float invl = 1.0f / lst[r];
#pragma unroll
    for (int ct = 0; ct < 4; ++ct) {
      const int cc = 16 * ct + lm;
      orow[m * Cn + cc] = g * acc[ct][r] * invl + xrow[m * Cn + cc];
    }
  }
}

// ------------------------------- launcher ----------------------------------

extern "C" void kernel_launch(void* const* d_in, const int* in_sizes, int n_in,
                              void* d_out, int out_size, void* d_ws, size_t ws_size,
                              hipStream_t stream) {
  const float* x     = (const float*)d_in[0];
  const float* Wq    = (const float*)d_in[1];
  const float* bq    = (const float*)d_in[2];
  const float* Wk    = (const float*)d_in[3];
  const float* bk    = (const float*)d_in[4];
  const float* Wv    = (const float*)d_in[5];
  const float* bv    = (const float*)d_in[6];
  const float* gamma = (const float*)d_in[7];
  float* out = (float*)d_out;

  const int NTOT = Bn * Tn * Cn; // 1,048,576
  // Workspace (f16): xh | qh | kh | vT | WtAll  ~= 8.4 MB
  _Float16* xh = (_Float16*)d_ws;
  _Float16* qh = xh + NTOT;
  _Float16* kh = qh + NTOT;
  _Float16* vT = kh + NTOT;
  _Float16* Wt = vT + NTOT;

  prep_convert_x<<<(NTOT / 4) / 256, 256, 0, stream>>>(x, xh, NTOT / 4);
  prep_transpose_w<<<(3 * Cn * Cn + 255) / 256, 256, 0, stream>>>(Wq, Wk, Wv, Wt);

  dim3 pgrid(Bn * Tn / 16, 3);
  proj_wmma<<<pgrid, 32, 0, stream>>>(xh, Wt, bq, bk, bv, qh, kh, vT);

  flash_attn<<<Bn * Tn / 16, 32, 0, stream>>>(kh, qh, vT, x, gamma, out);
}
